// CausalSelfAttention_58849641890655
// MI455X (gfx1250) — matrix operand order repaired and checked by reading, not compiled, not run on hardware
//
#include <hip/hip_runtime.h>

typedef __attribute__((ext_vector_type(16))) _Float16 v16h;
typedef __attribute__((ext_vector_type(8)))  _Float16 h8;
typedef __attribute__((ext_vector_type(8)))  float    v8f;
typedef unsigned int u32;
typedef __attribute__((ext_vector_type(4))) u32 u32x4;
typedef __attribute__((ext_vector_type(8))) u32 u32x8;

// ---------------------------------------------------------------------------
// WMMA fragment helpers (contiguous k order: element i holds k = 16*half + i, wave32)
// ---------------------------------------------------------------------------

// A-matrix 16x32 f16 fragment from row-major LDS [16 rows][>=32 cols], stride ld.
static __device__ __forceinline__ v16h ld_frag_a(const _Float16* base, int ld) {
  const int lane = threadIdx.x & 31;
  const int r = lane & 15, h = lane >> 4;
  const _Float16* p = base + r * ld + h * 16;
  v16h f;
#pragma unroll
  for (int i = 0; i < 8; ++i) f[i] = p[i];
#pragma unroll
  for (int i = 0; i < 8; ++i) f[i + 8] = p[8 + i];
  return f;
}

// B-matrix 32x16 f16 fragment. LDS holds B transposed: [16 n-rows][>=32 k].
static __device__ __forceinline__ v16h ld_frag_b(const _Float16* base, int ld) {
  const int lane = threadIdx.x & 31;
  const int n = lane & 15, h = lane >> 4;
  const _Float16* p = base + n * ld + h * 16;
  v16h f;
#pragma unroll
  for (int i = 0; i < 16; ++i) f[i] = p[i];
  return f;
}

static __device__ __forceinline__ v8f wmma_f16(v16h a, v16h b, v8f c) {
  return __builtin_amdgcn_wmma_f32_16x16x32_f16(false, a, false, b, (short)0, c,
                                                false, false);
}

// ---------------------------------------------------------------------------
// Tensor Data Mover: async 2D f16 tile load Global->LDS (CDNA5 ISA ch 8).
// D# group0/group1 per ISA 8.3/8.4; 2D tile so VADDR2/3 omitted (NULL).
// pad_interval code: 3 -> pad after 16 dwords; 4 -> after 32 dwords.
// pad_amount  code: 3 -> 4 dwords (8 f16).
// tensor dims == tile dims (tiles always fully in-bounds here).
// ---------------------------------------------------------------------------
static __device__ __forceinline__ void tdm_load_2d_f16(
    u32 lds_addr, const _Float16* gptr, u32 tile_d0, u32 tile_d1, u32 stride0,
    u32 pad_interval_code, u32 pad_amount_code) {
  unsigned long long ga = (unsigned long long)(__SIZE_TYPE__)gptr;
  u32x4 g0;
  g0[0] = 1u;                                            // count=1
  g0[1] = lds_addr;                                      // LDS byte address
  g0[2] = (u32)ga;                                       // global_addr[31:0]
  g0[3] = ((u32)(ga >> 32) & 0x01FFFFFFu) | (2u << 30);  // addr[56:32] | type=2
  u32x8 g1;
  g1[0] = (1u << 16)                 // data_size = 1 (2 bytes)
        | (1u << 20)                 // pad_enable
        | (pad_interval_code << 22)
        | (pad_amount_code << 25);
  g1[1] = (tile_d0 & 0xFFFFu) << 16;                     // tensor_dim0[15:0]
  g1[2] = (tile_d0 >> 16) | ((tile_d1 & 0xFFFFu) << 16); // dim0 hi | dim1 lo
  g1[3] = (tile_d1 >> 16) | (tile_d0 << 16);             // dim1 hi | tile_dim0
  g1[4] = tile_d1;                                       // tile_dim1 (tile_dim2=0)
  g1[5] = stride0;                                       // tensor_dim0_stride lo
  g1[6] = 0u;
  g1[7] = 0u;
  asm volatile("tensor_load_to_lds %0, %1" :: "s"(g0), "s"(g1) : "memory");
}

// ---------------------------------------------------------------------------
// GEMM: C[MxN] = A[MxK] * Bt[NxK]^T, A/Bt f16 row-major, f32 accum.
// Block 128x128, BK=64 (2 WMMA k-substeps), 256 threads (8 waves = 4M x 2N).
// Tiles staged by TDM with double buffering; wave 0 drives the DMA.
// ---------------------------------------------------------------------------
template <typename CT>
__global__ __launch_bounds__(256)
void gemm_tdm(const _Float16* __restrict__ A, const _Float16* __restrict__ Bt,
              CT* __restrict__ C, int K, int lda, int ldb, int ldc) {
  __shared__ _Float16 As[2][128][72];   // [m][k] (72 = 64 + TDM pad)
  __shared__ _Float16 Bs[2][128][72];   // [n][k]

  const int tid = threadIdx.x;
  const int m0 = blockIdx.y * 128;
  const int n0 = blockIdx.x * 128;
  const int w  = tid >> 5;
  const int wm = (w & 3) * 32;
  const int wn = (w >> 2) * 64;
  const int lane = tid & 31;
  const int half = lane >> 4, r = lane & 15;
  const int ksteps = K >> 6;

  v8f acc[2][4];
#pragma unroll
  for (int i = 0; i < 2; ++i)
#pragma unroll
    for (int j = 0; j < 4; ++j) acc[i][j] = v8f{};

  if (w == 0) {
    tdm_load_2d_f16((u32)(__SIZE_TYPE__)&As[0][0][0],
                    A + (size_t)m0 * lda, 64, 128, (u32)lda, 4, 3);
    tdm_load_2d_f16((u32)(__SIZE_TYPE__)&Bs[0][0][0],
                    Bt + (size_t)n0 * ldb, 64, 128, (u32)ldb, 4, 3);
  }

  for (int kt = 0; kt < ksteps; ++kt) {
    const int cur = kt & 1;
    if (w == 0) __builtin_amdgcn_s_wait_tensorcnt(0);
    __syncthreads();   // current tiles ready; previous buffer free
    if (w == 0 && (kt + 1) < ksteps) {
      const int k0 = (kt + 1) << 6;
      tdm_load_2d_f16((u32)(__SIZE_TYPE__)&As[cur ^ 1][0][0],
                      A + (size_t)m0 * lda + k0, 64, 128, (u32)lda, 4, 3);
      tdm_load_2d_f16((u32)(__SIZE_TYPE__)&Bs[cur ^ 1][0][0],
                      Bt + (size_t)n0 * ldb + k0, 64, 128, (u32)ldb, 4, 3);
    }

#pragma unroll
    for (int ks = 0; ks < 2; ++ks) {
      v16h a[2], b[4];
#pragma unroll
      for (int i = 0; i < 2; ++i)
        a[i] = ld_frag_a(&As[cur][wm + i * 16][ks * 32], 72);
#pragma unroll
      for (int j = 0; j < 4; ++j)
        b[j] = ld_frag_b(&Bs[cur][wn + j * 16][ks * 32], 72);
#pragma unroll
      for (int i = 0; i < 2; ++i)
#pragma unroll
        for (int j = 0; j < 4; ++j) acc[i][j] = wmma_f16(a[i], b[j], acc[i][j]);
    }
  }

  // epilogue: C layout row = e + 8*half, col = lane&15
#pragma unroll
  for (int i = 0; i < 2; ++i)
#pragma unroll
    for (int j = 0; j < 4; ++j) {
      CT* p = C + (size_t)(m0 + wm + i * 16 + half * 8) * ldc
                + (n0 + wn + j * 16 + r);
#pragma unroll
      for (int e = 0; e < 8; ++e) p[(size_t)e * ldc] = (CT)acc[i][j][e];
    }
}

// ---------------------------------------------------------------------------
// fp32 -> f16 convert (x) and fused transpose+convert (weights -> [N][K] f16)
// ---------------------------------------------------------------------------
__global__ __launch_bounds__(256)
void cvt_f16(const float* __restrict__ src, _Float16* __restrict__ dst) {
  int i = blockIdx.x * 256 + threadIdx.x;
  dst[i] = (_Float16)src[i];
}

__global__ __launch_bounds__(256)
void transpose_cvt(const float* __restrict__ W, _Float16* __restrict__ Wt,
                   int K, int N) {
  __shared__ float t[32][33];
  const int gx = blockIdx.x * 32;  // N
  const int gy = blockIdx.y * 32;  // K
  const int tx = threadIdx.x, ty = threadIdx.y;
#pragma unroll
  for (int j = 0; j < 32; j += 8)
    t[ty + j][tx] = W[(size_t)(gy + ty + j) * N + gx + tx];
  __syncthreads();
#pragma unroll
  for (int j = 0; j < 32; j += 8)
    Wt[(size_t)(gx + ty + j) * K + gy + tx] = (_Float16)t[tx][ty + j];
}

// ---------------------------------------------------------------------------
// RoPE (rotate_half) in place on f16 qkv [4096][3072].
// ---------------------------------------------------------------------------
__global__ __launch_bounds__(256)
void rope_kernel(_Float16* __restrict__ qkv) {
  int idx = blockIdx.x * 256 + threadIdx.x;   // T*H*32
  int t = idx >> 9;
  int rem = idx & 511;
  int h = rem >> 5;
  int i = rem & 31;
  float inv_freq = __powf(10000.0f, -(float)i * (1.0f / 32.0f));
  float ang = (float)t * inv_freq;
  float s, c;
  sincosf(ang, &s, &c);
  size_t base = (size_t)t * 3072 + h * 64 + i;
  {
    float a = (float)qkv[base], b = (float)qkv[base + 32];
    qkv[base]      = (_Float16)(a * c - b * s);
    qkv[base + 32] = (_Float16)(b * c + a * s);
  }
  {
    float a = (float)qkv[base + 1024], b = (float)qkv[base + 1024 + 32];
    qkv[base + 1024]      = (_Float16)(a * c - b * s);
    qkv[base + 1024 + 32] = (_Float16)(b * c + a * s);
  }
}

// ---------------------------------------------------------------------------
// Flash attention, causal. Grid (qblock=64, head=16), 128 threads = 4 waves.
// Q and K tiles staged by TDM (wave 0); V loaded manually (needs transpose).
// QK^T and P*V via v_wmma_f32_16x16x32_f16, online softmax in C-frag layout.
// ---------------------------------------------------------------------------
__global__ __launch_bounds__(128)
void flash_attn(const _Float16* __restrict__ qkv, _Float16* __restrict__ y) {
  __shared__ _Float16 Qs[64][72];      // [q][d]    (TDM pad -> stride 72)
  __shared__ _Float16 Ks[64][72];      // [key][d]  (TDM pad -> stride 72)
  __shared__ _Float16 Vt[64][72];      // [d][key]  (manual transpose)
  __shared__ _Float16 Ps[4][16][72];   // per-wave P staging

  const int tid = threadIdx.x;
  const int qb = blockIdx.x;
  const int h  = blockIdx.y;
  const int wv = tid >> 5;
  const int lane = tid & 31;
  const int half = lane >> 4, r = lane & 15;
  const int qoff = h * 64;

  // Q tile 64x64 via TDM (completion covered by first in-loop tensorcnt wait)
  if (wv == 0) {
    tdm_load_2d_f16((u32)(__SIZE_TYPE__)&Qs[0][0],
                    qkv + (size_t)(qb * 64) * 3072 + qoff, 64, 64, 3072, 4, 3);
  }

  float mst[8], lst[8];
  v8f o[4];
#pragma unroll
  for (int e = 0; e < 8; ++e) { mst[e] = -3.0e38f; lst[e] = 0.0f; }
#pragma unroll
  for (int f = 0; f < 4; ++f) o[f] = v8f{};

  for (int kb = 0; kb <= qb; ++kb) {
    __syncthreads();   // everyone done reading Ks/Vt from previous iteration
    if (wv == 0) {
      tdm_load_2d_f16((u32)(__SIZE_TYPE__)&Ks[0][0],
                      qkv + (size_t)(kb * 64) * 3072 + 1024 + qoff,
                      64, 64, 3072, 4, 3);
    }
    // V tile manual load + transpose scatter: 512 x 16B chunks, 4 per thread
#pragma unroll
    for (int i = 0; i < 4; ++i) {
      int idx = i * 128 + tid;
      int row = idx >> 3, c8 = (idx & 7) * 8;
      h8 vv = *(const h8*)(qkv + (size_t)(kb * 64 + row) * 3072 + 2048 + qoff + c8);
#pragma unroll
      for (int j = 0; j < 8; ++j) Vt[c8 + j][row] = vv[j];
    }
    if (wv == 0) __builtin_amdgcn_s_wait_tensorcnt(0);
    __syncthreads();   // K (and Q on first iter) + V ready

    // S = Q * K^T
    v8f s[4];
    {
      v16h aq0 = ld_frag_a(&Qs[wv * 16][0], 72);
      v16h aq1 = ld_frag_a(&Qs[wv * 16][32], 72);
#pragma unroll
      for (int j = 0; j < 4; ++j) {
        v8f sj = v8f{};
        sj = wmma_f16(aq0, ld_frag_b(&Ks[j * 16][0], 72), sj);
        sj = wmma_f16(aq1, ld_frag_b(&Ks[j * 16][32], 72), sj);
        s[j] = sj;
      }
    }

    const bool diag = (kb == qb);
#pragma unroll
    for (int j = 0; j < 4; ++j) {
      int key = kb * 64 + j * 16 + r;
#pragma unroll
      for (int e = 0; e < 8; ++e) {
        int qg = qb * 64 + wv * 16 + e + half * 8;
        float v = s[j][e] * 0.125f;           // 1/sqrt(64)
        if (diag && key > qg) v = -3.0e38f;
        s[j][e] = v;
      }
    }

    float alpha[8];
#pragma unroll
    for (int e = 0; e < 8; ++e) {
      float mx = fmaxf(fmaxf(s[0][e], s[1][e]), fmaxf(s[2][e], s[3][e]));
#pragma unroll
      for (int msk = 8; msk >= 1; msk >>= 1)
        mx = fmaxf(mx, __shfl_xor(mx, msk, 32));
      float mnew = fmaxf(mst[e], mx);
      alpha[e] = __expf(mst[e] - mnew);
      mst[e] = mnew;
      float rs = 0.0f;
#pragma unroll
      for (int j = 0; j < 4; ++j) {
        float p = __expf(s[j][e] - mnew);
        s[j][e] = p;
        rs += p;
      }
#pragma unroll
      for (int msk = 8; msk >= 1; msk >>= 1)
        rs += __shfl_xor(rs, msk, 32);
      lst[e] = lst[e] * alpha[e] + rs;
    }

#pragma unroll
    for (int f = 0; f < 4; ++f)
#pragma unroll
      for (int e = 0; e < 8; ++e) o[f][e] *= alpha[e];

#pragma unroll
    for (int j = 0; j < 4; ++j)
#pragma unroll
      for (int e = 0; e < 8; ++e)
        Ps[wv][e + half * 8][j * 16 + r] = (_Float16)s[j][e];
    __syncthreads();

    {
      v16h ap0 = ld_frag_a(&Ps[wv][0][0], 72);
      v16h ap1 = ld_frag_a(&Ps[wv][0][32], 72);
#pragma unroll
      for (int f = 0; f < 4; ++f) {
        o[f] = wmma_f16(ap0, ld_frag_b(&Vt[f * 16][0], 72), o[f]);
        o[f] = wmma_f16(ap1, ld_frag_b(&Vt[f * 16][32], 72), o[f]);
      }
    }
  }

#pragma unroll
  for (int f = 0; f < 4; ++f)
#pragma unroll
    for (int e = 0; e < 8; ++e) {
      int row = qb * 64 + wv * 16 + e + half * 8;
      int col = h * 64 + f * 16 + r;
      y[(size_t)row * 1024 + col] = (_Float16)(o[f][e] * (1.0f / lst[e]));
    }
}

// ---------------------------------------------------------------------------
// Launch. Workspace (all f16): qkv 24MB | xh 8MB | wqkvT 6MB | wprojT 2MB | y 8MB
// ---------------------------------------------------------------------------
extern "C" void kernel_launch(void* const* d_in, const int* in_sizes, int n_in,
                              void* d_out, int out_size, void* d_ws, size_t ws_size,
                              hipStream_t stream) {
  (void)in_sizes; (void)n_in; (void)out_size; (void)ws_size;
  const float* x      = (const float*)d_in[0];
  const float* w_qkv  = (const float*)d_in[1];
  const float* w_proj = (const float*)d_in[2];
  float* out = (float*)d_out;

  _Float16* qkv    = (_Float16*)d_ws;                  // [4096][3072]
  _Float16* xh     = qkv    + (size_t)4096 * 3072;     // [4096][1024]
  _Float16* wqkvT  = xh     + (size_t)4096 * 1024;     // [3072][1024]
  _Float16* wprojT = wqkvT  + (size_t)3072 * 1024;     // [1024][1024]
  _Float16* y      = wprojT + (size_t)1024 * 1024;     // [4096][1024]

  cvt_f16<<<(4096 * 1024) / 256, 256, 0, stream>>>(x, xh);
  transpose_cvt<<<dim3(3072 / 32, 1024 / 32), dim3(32, 8), 0, stream>>>(
      w_qkv, wqkvT, 1024, 3072);
  transpose_cvt<<<dim3(1024 / 32, 1024 / 32), dim3(32, 8), 0, stream>>>(
      w_proj, wprojT, 1024, 1024);

  // qkv = x @ w_qkv : M=4096, N=3072, K=1024
  gemm_tdm<_Float16><<<dim3(3072 / 128, 4096 / 128), 256, 0, stream>>>(
      xh, wqkvT, qkv, 1024, 1024, 1024, 3072);

  rope_kernel<<<(4096 * 16 * 32) / 256, 256, 0, stream>>>(qkv);

  flash_attn<<<dim3(4096 / 64, 16), 128, 0, stream>>>(qkv, y);

  // out = y @ w_proj : M=4096, N=1024, K=1024 (fp32 output)
  gemm_tdm<float><<<dim3(1024 / 128, 4096 / 128), 256, 0, stream>>>(
      y, wprojT, out, 1024, 1024, 1024, 1024);
}
